// LocalFeatureTransformer_40716289966124
// MI455X (gfx1250) — matrix operand order repaired and checked by reading, not compiled, not run on hardware
//
#include <hip/hip_runtime.h>
#include <hip/hip_bf16.h>
#include <cstdint>
#include <cstddef>

#define D_MODEL 256
#define NHEAD 8
#define HEAD_DIM 32
#define NB 4
#define LSEQ 4800
#define NLROWS (NB * LSEQ)          /* 19200 */
#define NLC (NLROWS * D_MODEL)      /* 4915200 */

typedef __attribute__((ext_vector_type(16))) __bf16 bf16x16;
typedef __attribute__((ext_vector_type(8)))  float  f32x8;
typedef __attribute__((ext_vector_type(4)))  int    v4i;

#if defined(__HIP_DEVICE_COMPILE__) && defined(__has_builtin)
#if __has_builtin(__builtin_amdgcn_global_load_async_to_lds_b128) && \
    __has_builtin(__builtin_amdgcn_s_wait_asynccnt)
#define HAVE_ASYNC_LDS 1
#endif
#endif
#ifndef HAVE_ASYNC_LDS
#define HAVE_ASYNC_LDS 0
#endif

#if HAVE_ASYNC_LDS
typedef __attribute__((address_space(1))) v4i glob_v4i;
typedef __attribute__((address_space(3))) v4i lds_v4i;
#endif

__device__ __forceinline__ __bf16 tobf(float f) {
  union { float f; uint32_t u; } un; un.f = f;
  uint32_t u = un.u;
  uint32_t r = (u + 0x7FFFu + ((u >> 16) & 1u)) >> 16;
  union { unsigned short s; __bf16 b; } o; o.s = (unsigned short)r;
  return o.b;
}

// 16-byte global -> LDS copy; async (ASYNCcnt) when available.
__device__ __forceinline__ void copy16_g2l(const __bf16* g, __bf16* l) {
#if HAVE_ASYNC_LDS
  __builtin_amdgcn_global_load_async_to_lds_b128(
      (glob_v4i*)g, (lds_v4i*)l, 0, 0);
#else
  *(uint4*)l = *(const uint4*)g;
#endif
}

__device__ __forceinline__ void async_fence() {
#if HAVE_ASYNC_LDS
  __builtin_amdgcn_s_wait_asynccnt(0);
#endif
}

// ---------------------------------------------------------------------------
// f32 -> bf16 conversion (count must be multiple of 256)
// ---------------------------------------------------------------------------
__global__ __launch_bounds__(256) void cvt_kernel(const float* __restrict__ X,
                                                  __bf16* __restrict__ Y) {
  size_t i = (size_t)blockIdx.x * 256 + threadIdx.x;
  Y[i] = tobf(X[i]);
}

// ---------------------------------------------------------------------------
// concat: Cat[r, 0:256] = X[r], Cat[r, 256:512] = Mg[r]   (bf16 out)
// ---------------------------------------------------------------------------
__global__ __launch_bounds__(256) void concat_kernel(const float* __restrict__ X,
                                                     const float* __restrict__ Mg,
                                                     __bf16* __restrict__ Cat) {
  size_t i = (size_t)blockIdx.x * 256 + threadIdx.x;
  size_t r = i >> 9;
  int c = (int)(i & 511);
  float v = (c < 256) ? X[r * 256 + c] : Mg[r * 256 + (c - 256)];
  Cat[i] = tobf(v);
}

// ---------------------------------------------------------------------------
// WMMA bf16 GEMM:  C[M,N] = A[M,K] @ B[K,N]
//   OP: 0 = f32 out, 1 = elu(x)+1 f32 out, 2 = x*scale f32 out,
//       3 = relu -> bf16 out
// Block 256 thr (8 waves). Block tile 128x128, wave tile 32x64 (2x4 frags).
// K staged in 32-deep chunks, double-buffered LDS; A tile via async-to-LDS,
// B tile register-staged with in-thread transpose (n-major in LDS).
// Requires M%128==0, N%128==0, K%32==0.
// ---------------------------------------------------------------------------
__device__ __forceinline__ bf16x16 load_frag_a(const __bf16* p0, int lane) {
  // A 16x32 bf16 frag: lane<16 -> m=lane, k in {0..7,16..23};
  //                    lane>=16 -> m=lane-16, k in {8..15,24..31}
  const __bf16* p = p0 + (size_t)(lane & 15) * 32 + ((lane < 16) ? 0 : 8);
  bf16x16 f;
#pragma unroll
  for (int i = 0; i < 8; i++) { f[i] = p[i]; f[8 + i] = p[16 + i]; }
  return f;
}

__device__ __forceinline__ bf16x16 load_frag_b(const __bf16* p0, int lane) {
  // B 32x16 bf16 frag from n-major LDS (Bt[n][k]):
  //   lane<16 -> n=lane, k in {0..7,16..23} ; lane>=16 -> n=lane-16, k in {8..15,24..31}
  const __bf16* p = p0 + (size_t)(lane & 15) * 32 + ((lane < 16) ? 0 : 8);
  bf16x16 f;
#pragma unroll
  for (int i = 0; i < 16; i++) f[i] = p[(i < 8) ? i : i + 8];
  return f;
}

template <int OP>
__global__ __launch_bounds__(256) void gemm_bf16(const __bf16* __restrict__ A,
                                                 const __bf16* __restrict__ B,
                                                 void* __restrict__ Cp,
                                                 int M, int Nn, int K, float scale) {
  __shared__ __align__(16) __bf16 As[2][128][32];
  __shared__ __align__(16) __bf16 Bt[2][128][32];

  const int tid = threadIdx.x;
  const int lane = tid & 31;
  const int wid = tid >> 5;
  const int wm = wid >> 1;        // 0..3 : 32-row band
  const int wn = wid & 1;         // 0..1 : 64-col band
  const int m0 = blockIdx.y * 128;
  const int n0 = blockIdx.x * 128;

  // A staging: 128x32 bf16 = 512 x 16B chunks, 2 per thread
  // B staging: 32x128 bf16 = 512 x 16B chunks, 2 per thread (transposed store)
  __bf16 br[2][8];

  auto stage_a = [&](int buf, int k0) {
#pragma unroll
    for (int it = 0; it < 2; it++) {
      int idx = tid + it * 256;                 // 0..511
      int r = idx >> 2;
      int cc = (idx & 3) * 8;
      copy16_g2l(A + (size_t)(m0 + r) * K + k0 + cc, &As[buf][r][cc]);
    }
  };
  auto load_b_regs = [&](int k0) {
#pragma unroll
    for (int it = 0; it < 2; it++) {
      int idx = tid + it * 256;                 // 0..511
      int kk = idx >> 4;                        // 0..31
      int nn = (idx & 15) * 8;                  // 0..120
      *(uint4*)br[it] = *(const uint4*)(B + (size_t)(k0 + kk) * Nn + n0 + nn);
    }
  };
  auto store_b = [&](int buf) {
#pragma unroll
    for (int it = 0; it < 2; it++) {
      int idx = tid + it * 256;
      int kk = idx >> 4;
      int nn = (idx & 15) * 8;
#pragma unroll
      for (int j = 0; j < 8; j++) Bt[buf][nn + j][kk] = br[it][j];
    }
  };

  f32x8 acc[2][4] = {};

  // Prologue: fill buffer 0
  stage_a(0, 0);
  load_b_regs(0);
  store_b(0);
  async_fence();
  __syncthreads();

  const int nk = K / 32;
  for (int kt = 0; kt < nk; kt++) {
    const int cur = kt & 1;
    const int nxt = cur ^ 1;
    const bool more = (kt + 1) < nk;
    if (more) {
      stage_a(nxt, (kt + 1) * 32);              // async global -> LDS (next)
      load_b_regs((kt + 1) * 32);               // global -> regs (next)
    }

    bf16x16 af[2], bfv[4];
#pragma unroll
    for (int i = 0; i < 2; i++)
      af[i] = load_frag_a(&As[cur][wm * 32 + i * 16][0], lane);
#pragma unroll
    for (int j = 0; j < 4; j++)
      bfv[j] = load_frag_b(&Bt[cur][wn * 64 + j * 16][0], lane);

#pragma unroll
    for (int i = 0; i < 2; i++)
#pragma unroll
      for (int j = 0; j < 4; j++)
        acc[i][j] = __builtin_amdgcn_wmma_f32_16x16x32_bf16(
            false, af[i], false, bfv[j], (short)0, acc[i][j], false, false);

    if (more) {
      store_b(nxt);                             // regs -> LDS (next, transposed)
      async_fence();                            // this wave's async A landed
    }
    __syncthreads();
  }

  // Epilogue: C/D 16x16 f32 layout: vgpr r -> (M=r, N=lane) lane<16,
  //           (M=8+r, N=lane-16) lane>=16
  const int col0 = n0 + wn * 64 + (lane & 15);
  const int row0 = m0 + wm * 32 + ((lane < 16) ? 0 : 8);
#pragma unroll
  for (int i = 0; i < 2; i++)
#pragma unroll
    for (int j = 0; j < 4; j++)
#pragma unroll
      for (int r = 0; r < 8; r++) {
        float v = acc[i][j][r];
        int row = row0 + i * 16 + r;
        int c = col0 + j * 16;
        if (OP == 1) v = (v > 0.f) ? (v + 1.f) : __expf(v);   // elu(x)+1
        else if (OP == 2) v = v * scale;
        if (OP == 3) {
          float rv = v > 0.f ? v : 0.f;
          ((__bf16*)Cp)[(size_t)row * Nn + c] = tobf(rv);
        } else {
          ((float*)Cp)[(size_t)row * Nn + c] = v;
        }
      }
}

// ---------------------------------------------------------------------------
// KV reduction: KV[n,h,d,v] = sum_s K[n,s,h*32+d] * V[n,s,h*32+v]
//               Ksum[n,h,d] = sum_s K[n,s,h*32+d]
// ---------------------------------------------------------------------------
__global__ __launch_bounds__(256) void kv_kernel(const float* __restrict__ K,
                                                 const float* __restrict__ V,
                                                 float* __restrict__ KV,
                                                 float* __restrict__ Ksum) {
  const int nh = blockIdx.x;
  const int n = nh >> 3;
  const int h = nh & 7;
  __shared__ float Ks[16][32];
  __shared__ float Vs[16][32];
  const int tid = threadIdx.x;
  const int d = tid >> 3;
  const int v0 = (tid & 7) * 4;
  float acc0 = 0.f, acc1 = 0.f, acc2 = 0.f, acc3 = 0.f, ks = 0.f;
  const float* Kb = K + (size_t)n * LSEQ * D_MODEL + h * 32;
  const float* Vb = V + (size_t)n * LSEQ * D_MODEL + h * 32;
  for (int s0 = 0; s0 < LSEQ; s0 += 16) {
    for (int i = tid; i < 512; i += 256) {
      int rr = i >> 5, cc = i & 31;
      Ks[rr][cc] = Kb[(size_t)(s0 + rr) * D_MODEL + cc];
      Vs[rr][cc] = Vb[(size_t)(s0 + rr) * D_MODEL + cc];
    }
    __syncthreads();
#pragma unroll
    for (int ss = 0; ss < 16; ss++) {
      float kv = Ks[ss][d];
      if ((tid & 7) == 0) ks += kv;
      acc0 += kv * Vs[ss][v0 + 0];
      acc1 += kv * Vs[ss][v0 + 1];
      acc2 += kv * Vs[ss][v0 + 2];
      acc3 += kv * Vs[ss][v0 + 3];
    }
    __syncthreads();
  }
  float* o = KV + (size_t)nh * 1024 + d * 32 + v0;
  o[0] = acc0; o[1] = acc1; o[2] = acc2; o[3] = acc3;
  if ((tid & 7) == 0) Ksum[nh * 32 + d] = ks;
}

// ---------------------------------------------------------------------------
// msg[n,l,h*32+v] = S * Z(n,l,h) * sum_d Q[n,l,h*32+d]*KV[n,h,d,v]
// ---------------------------------------------------------------------------
__global__ __launch_bounds__(256) void msg_kernel(const float* __restrict__ Q,
                                                  const float* __restrict__ KV,
                                                  const float* __restrict__ Ksum,
                                                  float* __restrict__ Msg) {
  const int b = blockIdx.x;
  const int n = b / (LSEQ / 16);
  const int l0 = (b % (LSEQ / 16)) * 16;
  __shared__ float KVs[NHEAD][32][32];
  __shared__ float Kss[NHEAD][32];
  __shared__ float Qs[256];
  const int tid = threadIdx.x;
  const float* KVsrc = KV + (size_t)n * NHEAD * 1024;
  for (int i = tid; i < NHEAD * 1024; i += 256) ((float*)KVs)[i] = KVsrc[i];
  ((float*)Kss)[tid] = Ksum[n * 256 + tid];
  __syncthreads();
  const int h = tid >> 5, v = tid & 31;
  for (int li = 0; li < 16; li++) {
    size_t row = ((size_t)n * LSEQ + l0 + li) * D_MODEL;
    Qs[tid] = Q[row + tid];
    __syncthreads();
    float zden = 0.f, o = 0.f;
#pragma unroll
    for (int d = 0; d < 32; d++) {
      float qd = Qs[h * 32 + d];
      zden += qd * Kss[h][d];
      o += qd * KVs[h][d][v];
    }
    float z = 1.0f / (zden + 1e-6f);
    Msg[row + tid] = o * z * (float)LSEQ;
    __syncthreads();
  }
}

// ---------------------------------------------------------------------------
// LayerNorm over C=256: Y = (res ? res + ln(X) : ln(X)) ; one block per row
// ---------------------------------------------------------------------------
__global__ __launch_bounds__(256) void ln_kernel(const float* __restrict__ X,
                                                 const float* __restrict__ g,
                                                 const float* __restrict__ b,
                                                 const float* __restrict__ res,
                                                 float* __restrict__ Y) {
  __shared__ float sh[256];
  const size_t row = (size_t)blockIdx.x * 256;
  const int tid = threadIdx.x;
  float x = X[row + tid];
  sh[tid] = x;
  __syncthreads();
  for (int off = 128; off > 0; off >>= 1) {
    if (tid < off) sh[tid] += sh[tid + off];
    __syncthreads();
  }
  float mu = sh[0] * (1.0f / 256.0f);
  __syncthreads();
  float d = x - mu;
  sh[tid] = d * d;
  __syncthreads();
  for (int off = 128; off > 0; off >>= 1) {
    if (tid < off) sh[tid] += sh[tid + off];
    __syncthreads();
  }
  float var = sh[0] * (1.0f / 256.0f);
  float y = d * rsqrtf(var + 1e-5f) * g[tid] + b[tid];
  if (res) y += res[row + tid];
  Y[row + tid] = y;
}

// ---------------------------------------------------------------------------
// Host-side orchestration
// ---------------------------------------------------------------------------
namespace {

struct Scratch {
  __bf16 *WqB, *WkB, *WvB, *WmB, *W1B, *W2B;
  __bf16 *xb, *sb;
  float *Qf, *Kf, *Vf, *tmp;
  float *KVb, *Ksb;
};

static void run_encoder(const float* x, const float* src, float* dst,
                        const Scratch& S,
                        const __bf16* wq, const __bf16* wk, const __bf16* wv,
                        const __bf16* wm, const __bf16* w1, const __bf16* w2,
                        const float* G1, const float* B1,
                        const float* G2, const float* B2,
                        hipStream_t stream) {
  const dim3 blk(256);
  const dim3 gN256(D_MODEL / 128, NLROWS / 128);
  const dim3 gN512(512 / 128, NLROWS / 128);

  __bf16* mb = S.sb;               // source bf16 free after K/V gemms
  float* Mf = S.Kf;                // K free after kv_kernel
  float* MsgW = S.Vf;              // V free after kv_kernel
  __bf16* catb = (__bf16*)S.Kf;    // Mf free after cvt to mb
  __bf16* h1b = (__bf16*)S.Vf;     // MsgW free after concat

  cvt_kernel<<<NLC / 256, blk, 0, stream>>>(x, S.xb);
  cvt_kernel<<<NLC / 256, blk, 0, stream>>>(src, S.sb);

  gemm_bf16<1><<<gN256, blk, 0, stream>>>(S.xb, wq, S.Qf, NLROWS, D_MODEL, D_MODEL, 0.f);
  gemm_bf16<1><<<gN256, blk, 0, stream>>>(S.sb, wk, S.Kf, NLROWS, D_MODEL, D_MODEL, 0.f);
  gemm_bf16<2><<<gN256, blk, 0, stream>>>(S.sb, wv, S.Vf, NLROWS, D_MODEL, D_MODEL,
                                          1.0f / (float)LSEQ);

  kv_kernel<<<NB * NHEAD, blk, 0, stream>>>(S.Kf, S.Vf, S.KVb, S.Ksb);
  msg_kernel<<<NB * (LSEQ / 16), blk, 0, stream>>>(S.Qf, S.KVb, S.Ksb, Mf);

  cvt_kernel<<<NLC / 256, blk, 0, stream>>>(Mf, mb);
  gemm_bf16<0><<<gN256, blk, 0, stream>>>(mb, wm, MsgW, NLROWS, D_MODEL, D_MODEL, 0.f);
  ln_kernel<<<NLROWS, blk, 0, stream>>>(MsgW, G1, B1, nullptr, MsgW);

  concat_kernel<<<(NLROWS * 512) / 256, blk, 0, stream>>>(x, MsgW, catb);
  gemm_bf16<3><<<gN512, blk, 0, stream>>>(catb, w1, h1b, NLROWS, 512, 512, 0.f);
  gemm_bf16<0><<<gN256, blk, 0, stream>>>(h1b, w2, S.Qf, NLROWS, D_MODEL, 512, 0.f);
  ln_kernel<<<NLROWS, blk, 0, stream>>>(S.Qf, G2, B2, x, dst);
}

}  // namespace

extern "C" void kernel_launch(void* const* d_in, const int* in_sizes, int n_in,
                              void* d_out, int out_size, void* d_ws, size_t ws_size,
                              hipStream_t stream) {
  (void)in_sizes; (void)n_in; (void)out_size; (void)ws_size;

  const float* feat0 = (const float*)d_in[0];
  const float* feat1 = (const float*)d_in[1];
  const float* Wq = (const float*)d_in[2];
  const float* Wk = (const float*)d_in[3];
  const float* Wv = (const float*)d_in[4];
  const float* Wm = (const float*)d_in[5];
  const float* W1 = (const float*)d_in[6];
  const float* W2 = (const float*)d_in[7];
  const float* g1 = (const float*)d_in[8];
  const float* b1 = (const float*)d_in[9];
  const float* g2 = (const float*)d_in[10];
  const float* b2 = (const float*)d_in[11];

  float* out0 = (float*)d_out;
  float* out1 = out0 + (size_t)NLC;

  // Carve workspace
  char* p = (char*)d_ws;
  auto alloc = [&](size_t bytes) -> void* {
    void* r = (void*)p;
    p += (bytes + 255) & ~(size_t)255;
    return r;
  };
  Scratch S;
  S.WqB = (__bf16*)alloc((size_t)8 * 256 * 256 * 2);
  S.WkB = (__bf16*)alloc((size_t)8 * 256 * 256 * 2);
  S.WvB = (__bf16*)alloc((size_t)8 * 256 * 256 * 2);
  S.WmB = (__bf16*)alloc((size_t)8 * 256 * 256 * 2);
  S.W1B = (__bf16*)alloc((size_t)8 * 512 * 512 * 2);
  S.W2B = (__bf16*)alloc((size_t)8 * 512 * 256 * 2);
  S.xb  = (__bf16*)alloc((size_t)NLC * 2);
  S.sb  = (__bf16*)alloc((size_t)NLC * 2);
  S.Qf  = (float*)alloc((size_t)NLC * 4);
  S.Kf  = (float*)alloc((size_t)NLROWS * 512 * 4);  // also catb (bf16, NL*512)
  S.Vf  = (float*)alloc((size_t)NLROWS * 512 * 4);  // also h1b  (bf16, NL*512)
  S.tmp = (float*)alloc((size_t)NLC * 4);
  S.KVb = (float*)alloc((size_t)NB * NHEAD * 32 * 32 * 4);
  S.Ksb = (float*)alloc((size_t)NB * NHEAD * 32 * 4);

  // Convert all weights to bf16 once per call
  cvt_kernel<<<(8 * 256 * 256) / 256, 256, 0, stream>>>(Wq, S.WqB);
  cvt_kernel<<<(8 * 256 * 256) / 256, 256, 0, stream>>>(Wk, S.WkB);
  cvt_kernel<<<(8 * 256 * 256) / 256, 256, 0, stream>>>(Wv, S.WvB);
  cvt_kernel<<<(8 * 256 * 256) / 256, 256, 0, stream>>>(Wm, S.WmB);
  cvt_kernel<<<(8 * 512 * 512) / 256, 256, 0, stream>>>(W1, S.W1B);
  cvt_kernel<<<(8 * 512 * 256) / 256, 256, 0, stream>>>(W2, S.W2B);

  // Residual stream lives in d_out
  (void)hipMemcpyAsync(out0, feat0, (size_t)NLC * 4, hipMemcpyDeviceToDevice, stream);
  (void)hipMemcpyAsync(out1, feat1, (size_t)NLC * 4, hipMemcpyDeviceToDevice, stream);

  static const int is_self[8] = {1, 0, 1, 0, 1, 0, 1, 0};
  for (int i = 0; i < 8; i++) {
    const __bf16* wq = S.WqB + (size_t)i * 256 * 256;
    const __bf16* wk = S.WkB + (size_t)i * 256 * 256;
    const __bf16* wv = S.WvB + (size_t)i * 256 * 256;
    const __bf16* wm = S.WmB + (size_t)i * 256 * 256;
    const __bf16* w1 = S.W1B + (size_t)i * 512 * 512;
    const __bf16* w2 = S.W2B + (size_t)i * 512 * 256;
    const float* G1 = g1 + i * 256;
    const float* B1 = b1 + i * 256;
    const float* G2 = g2 + i * 256;
    const float* B2 = b2 + i * 256;

    if (is_self[i]) {
      run_encoder(out0, out0, out0, S, wq, wk, wv, wm, w1, w2, G1, B1, G2, B2, stream);
      run_encoder(out1, out1, out1, S, wq, wk, wv, wm, w1, w2, G1, B1, G2, B2, stream);
    } else {
      run_encoder(out0, out1, S.tmp, S, wq, wk, wv, wm, w1, w2, G1, B1, G2, B2, stream);
      run_encoder(out1, out0, out1, S, wq, wk, wv, wm, w1, w2, G1, B1, G2, B2, stream);
      (void)hipMemcpyAsync(out0, S.tmp, (size_t)NLC * 4, hipMemcpyDeviceToDevice, stream);
    }
  }
}
